// AttentionLikeModel_75531294867774
// MI455X (gfx1250) — matrix operand order repaired and checked by reading, not compiled, not run on hardware
//
#include <hip/hip_runtime.h>

#define DIM 1024
#define SEQ 2048
#define NB  2
#define NH  16
#define HD  64
#define TOK (NB*SEQ)   // 4096

typedef __attribute__((ext_vector_type(16))) _Float16 v16h;
typedef __attribute__((ext_vector_type(8)))  _Float16 v8h;
typedef __attribute__((ext_vector_type(8)))  float    v8f;

union F16F { v16h v; v8h h[2]; _Float16 e[16]; };

static __device__ __forceinline__ v8h ldh8(const _Float16* p) {
  return *(const v8h*)p;
}
static __device__ __forceinline__ v8f wmma_f16(v16h a, v16h b, v8f c) {
  // D = A(16x32 f16) * B(32x16 f16) + C(16x16 f32)
  return __builtin_amdgcn_wmma_f32_16x16x32_f16(false, a, false, b, (short)0, c, false, false);
}

// ---------------- prep: fp32 -> fp16 conversions / weight transposes ----------------
__global__ void k_cvt_x(const float* __restrict__ x, _Float16* __restrict__ xh, int n) {
  int i = blockIdx.x * blockDim.x + threadIdx.x;
  if (i < n) xh[i] = (_Float16)x[i];
}
// W [K,N] row-major f32 -> Wt [N,K] row-major f16
__global__ void k_transpose(const float* __restrict__ W, _Float16* __restrict__ Wt, int K, int N) {
  int i = blockIdx.x * blockDim.x + threadIdx.x;
  if (i < K * N) {
    int k = i / N, n = i % N;
    Wt[(size_t)n * K + k] = (_Float16)W[i];
  }
}

// ---------------- pass 1: qkv = x @ W_qkv + b; 32x64 tile per wave ----------------
__global__ void __launch_bounds__(128) k_qkv(const _Float16* __restrict__ xh,
                                             const _Float16* __restrict__ wt,
                                             const float* __restrict__ bias,
                                             _Float16* __restrict__ qh,
                                             _Float16* __restrict__ kh,
                                             _Float16* __restrict__ vt) {
  const int lane = threadIdx.x & 31;
  const int wave = threadIdx.x >> 5;
  const int tile = blockIdx.x * 4 + wave;   // (TOK/32)*(3*DIM/64) = 6144 tiles
  const int NTN  = (3 * DIM) / 64;          // 48
  const int rt = tile / NTN, ct = tile % NTN;
  const int m0 = rt * 32, n0 = ct * 64;
  const int m  = lane & 15, hb = lane >> 4, ab = hb * 8;
  const int ncol = n0 + (lane & 15);

  v8f z = {};
  v8f acc[2][4] = {{z, z, z, z}, {z, z, z, z}};

  const _Float16* arow0 = xh + (size_t)(m0 + m) * DIM;
  const _Float16* arow1 = arow0 + (size_t)16 * DIM;
  for (int k0 = 0; k0 < DIM; k0 += 32) {
    F16F a0, a1;
    a0.h[0] = ldh8(arow0 + k0 + ab); a0.h[1] = ldh8(arow0 + k0 + 16 + ab);
    a1.h[0] = ldh8(arow1 + k0 + ab); a1.h[1] = ldh8(arow1 + k0 + 16 + ab);
#pragma unroll
    for (int nt = 0; nt < 4; ++nt) {
      const _Float16* bp = wt + (size_t)(ncol + nt * 16) * DIM + k0 + hb * 8;
      F16F b; b.h[0] = ldh8(bp); b.h[1] = ldh8(bp + 16);
      acc[0][nt] = wmma_f16(a0.v, b.v, acc[0][nt]);   // B reused across both row tiles
      acc[1][nt] = wmma_f16(a1.v, b.v, acc[1][nt]);
    }
  }
#pragma unroll
  for (int nt = 0; nt < 4; ++nt) {
    int ng = ncol + nt * 16;
    float bv = bias[ng];
    int sec = ng / DIM, c = ng % DIM;
    int h = c / HD, d = c % HD;
#pragma unroll
    for (int mt = 0; mt < 2; ++mt) {
#pragma unroll
      for (int r = 0; r < 8; ++r) {
        int row = m0 + mt * 16 + r + hb * 8;   // token index (2048 % 32 == 0, no batch straddle)
        int b_ = row / SEQ, s = row % SEQ;
        float val = acc[mt][nt][r] + bv;
        size_t bh = (size_t)(b_ * NH + h);
        if (sec == 0)      qh[(bh * SEQ + s) * HD + d] = (_Float16)val;
        else if (sec == 1) kh[(bh * SEQ + s) * HD + d] = (_Float16)val;
        else               vt[(bh * HD + d) * SEQ + s] = (_Float16)val;
      }
    }
  }
}

// ---------------- pass 2: flash attention, 32 q-rows per wave ----------------
__global__ void __launch_bounds__(128) k_attn(const _Float16* __restrict__ qh,
                                              const _Float16* __restrict__ kh,
                                              const _Float16* __restrict__ vt,
                                              _Float16* __restrict__ oh) {
  __shared__ _Float16 plds[4][2][16][40];  // per-wave, per-qtile 16x32 P, padded rows
  const int lane = threadIdx.x & 31;
  const int wave = threadIdx.x >> 5;
  const int bh   = blockIdx.x;             // 0..31 (b*16+h)
  const int q0   = blockIdx.y * 128 + wave * 32;
  const int m = lane & 15, hb = lane >> 4, ab = hb * 8;

  const size_t base = (size_t)bh * SEQ * HD;   // same stride for qh/kh/vt

  F16F qa[2][2];
#pragma unroll
  for (int mt = 0; mt < 2; ++mt) {
    const _Float16* qrow = qh + base + (size_t)(q0 + mt * 16 + m) * HD;
    qa[mt][0].h[0] = ldh8(qrow + ab);      qa[mt][0].h[1] = ldh8(qrow + 16 + ab);
    qa[mt][1].h[0] = ldh8(qrow + 32 + ab); qa[mt][1].h[1] = ldh8(qrow + 48 + ab);
  }

  float mi[2][8], li[2][8];
  v8f z = {};
  v8f acc[2][4] = {{z, z, z, z}, {z, z, z, z}};
#pragma unroll
  for (int mt = 0; mt < 2; ++mt)
#pragma unroll
    for (int r = 0; r < 8; ++r) { mi[mt][r] = -__builtin_inff(); li[mt][r] = 0.f; }

  for (int kc = 0; kc < SEQ; kc += 32) {
    v8f s[2][2];
#pragma unroll
    for (int nt = 0; nt < 2; ++nt) {
      const _Float16* kp = kh + base + (size_t)(kc + nt * 16 + (lane & 15)) * HD + hb * 8;
      F16F b0, b1;
      b0.h[0] = ldh8(kp);      b0.h[1] = ldh8(kp + 16);
      b1.h[0] = ldh8(kp + 32); b1.h[1] = ldh8(kp + 48);
#pragma unroll
      for (int mt = 0; mt < 2; ++mt) {     // K fragments reused across both q tiles
        s[mt][nt] = wmma_f16(qa[mt][0].v, b0.v, z);
        s[mt][nt] = wmma_f16(qa[mt][1].v, b1.v, s[mt][nt]);
      }
      if (kc + 32 < SEQ) __builtin_prefetch(kp + 32 * HD, 0, 1);
    }
    const float scale = 0.125f;            // HD^-0.5
#pragma unroll
    for (int mt = 0; mt < 2; ++mt) {
      float p0[8], p1[8], alpha[8];
#pragma unroll
      for (int r = 0; r < 8; ++r) {
        float a0 = s[mt][0][r] * scale, a1 = s[mt][1][r] * scale;
        float rm = fmaxf(a0, a1);          // row reduce within 16-lane half
        rm = fmaxf(rm, __shfl_xor(rm, 1, 32));
        rm = fmaxf(rm, __shfl_xor(rm, 2, 32));
        rm = fmaxf(rm, __shfl_xor(rm, 4, 32));
        rm = fmaxf(rm, __shfl_xor(rm, 8, 32));
        float mn = fmaxf(mi[mt][r], rm);
        alpha[r] = __expf(mi[mt][r] - mn);
        p0[r] = __expf(a0 - mn);
        p1[r] = __expf(a1 - mn);
        float rs = p0[r] + p1[r];
        rs += __shfl_xor(rs, 1, 32);
        rs += __shfl_xor(rs, 2, 32);
        rs += __shfl_xor(rs, 4, 32);
        rs += __shfl_xor(rs, 8, 32);
        li[mt][r] = li[mt][r] * alpha[r] + rs;
        mi[mt][r] = mn;
      }
#pragma unroll
      for (int dt = 0; dt < 4; ++dt)
#pragma unroll
        for (int r = 0; r < 8; ++r)
          acc[mt][dt][r] *= alpha[r];
      // C-layout -> row-major P tile in LDS (per-wave region; DS in-order per wave)
#pragma unroll
      for (int r = 0; r < 8; ++r) {
        plds[wave][mt][r + hb * 8][lane & 15]        = (_Float16)p0[r];
        plds[wave][mt][r + hb * 8][16 + (lane & 15)] = (_Float16)p1[r];
      }
    }
    asm volatile("s_wait_dscnt 0" ::: "memory");
    F16F pa[2];                             // reload P as A-fragments (K = 32)
#pragma unroll
    for (int mt = 0; mt < 2; ++mt)
#pragma unroll
      for (int j = 0; j < 8; ++j) {
        pa[mt].e[j]     = plds[wave][mt][m][ab + j];
        pa[mt].e[8 + j] = plds[wave][mt][m][16 + ab + j];
      }
    // O += P @ V  (V fragments reused across both q tiles)
#pragma unroll
    for (int dt = 0; dt < 4; ++dt) {
      const _Float16* vp = vt + base + (size_t)(dt * 16 + (lane & 15)) * SEQ + kc + hb * 8;
      F16F b; b.h[0] = ldh8(vp); b.h[1] = ldh8(vp + 16);
      acc[0][dt] = wmma_f16(pa[0].v, b.v, acc[0][dt]);
      acc[1][dt] = wmma_f16(pa[1].v, b.v, acc[1][dt]);
    }
    asm volatile("s_wait_dscnt 0" ::: "memory");  // P reads done before next-iter overwrite
  }
  const int b_ = bh >> 4, h = bh & 15;
#pragma unroll
  for (int mt = 0; mt < 2; ++mt)
#pragma unroll
    for (int dt = 0; dt < 4; ++dt) {
      int col = h * HD + dt * 16 + (lane & 15);
#pragma unroll
      for (int r = 0; r < 8; ++r) {
        int tok = b_ * SEQ + q0 + mt * 16 + r + hb * 8;
        oh[(size_t)tok * DIM + col] = (_Float16)(acc[mt][dt][r] / li[mt][r]);
      }
    }
}

// ---------------- pass 3: out = O @ W_out + b_out; 32x64 tile per wave ----------------
__global__ void __launch_bounds__(128) k_proj(const _Float16* __restrict__ oh,
                                              const _Float16* __restrict__ wt,
                                              const float* __restrict__ bias,
                                              float* __restrict__ out) {
  const int lane = threadIdx.x & 31;
  const int wave = threadIdx.x >> 5;
  const int tile = blockIdx.x * 4 + wave;   // (TOK/32)*(DIM/64) = 2048 tiles
  const int NTN  = DIM / 64;                // 16
  const int rt = tile / NTN, ct = tile % NTN;
  const int m0 = rt * 32, n0 = ct * 64;
  const int m = lane & 15, hb = lane >> 4, ab = hb * 8;
  const int ncol = n0 + (lane & 15);

  v8f z = {};
  v8f acc[2][4] = {{z, z, z, z}, {z, z, z, z}};
  const _Float16* arow0 = oh + (size_t)(m0 + m) * DIM;
  const _Float16* arow1 = arow0 + (size_t)16 * DIM;
  for (int k0 = 0; k0 < DIM; k0 += 32) {
    F16F a0, a1;
    a0.h[0] = ldh8(arow0 + k0 + ab); a0.h[1] = ldh8(arow0 + k0 + 16 + ab);
    a1.h[0] = ldh8(arow1 + k0 + ab); a1.h[1] = ldh8(arow1 + k0 + 16 + ab);
#pragma unroll
    for (int nt = 0; nt < 4; ++nt) {
      const _Float16* bp = wt + (size_t)(ncol + nt * 16) * DIM + k0 + hb * 8;
      F16F b; b.h[0] = ldh8(bp); b.h[1] = ldh8(bp + 16);
      acc[0][nt] = wmma_f16(a0.v, b.v, acc[0][nt]);
      acc[1][nt] = wmma_f16(a1.v, b.v, acc[1][nt]);
    }
  }
#pragma unroll
  for (int nt = 0; nt < 4; ++nt) {
    int ng = ncol + nt * 16;
    float bv = bias[ng];
#pragma unroll
    for (int mt = 0; mt < 2; ++mt)
#pragma unroll
      for (int r = 0; r < 8; ++r) {
        int row = m0 + mt * 16 + r + hb * 8;
        out[(size_t)row * DIM + ng] = acc[mt][nt][r] + bv;
      }
  }
}

// ---------------- host ----------------
extern "C" void kernel_launch(void* const* d_in, const int* in_sizes, int n_in,
                              void* d_out, int out_size, void* d_ws, size_t ws_size,
                              hipStream_t stream) {
  (void)in_sizes; (void)n_in; (void)out_size; (void)ws_size;
  const float* x    = (const float*)d_in[0];
  const float* Wqkv = (const float*)d_in[1];
  const float* bqkv = (const float*)d_in[2];
  const float* Wout = (const float*)d_in[3];
  const float* bout = (const float*)d_in[4];

  char* ws = (char*)d_ws;
  _Float16* xh  = (_Float16*)(ws);                      //  8 MiB  x in f16
  _Float16* wqt = (_Float16*)(ws + ( 8ull << 20));      //  6 MiB  W_qkv^T f16
  _Float16* wot = (_Float16*)(ws + (14ull << 20));      //  2 MiB  W_out^T  f16
  _Float16* qh  = (_Float16*)(ws + (16ull << 20));      //  8 MiB  Q [B,H,S,hd]
  _Float16* kh  = (_Float16*)(ws + (24ull << 20));      //  8 MiB  K [B,H,S,hd]
  _Float16* vt  = (_Float16*)(ws + (32ull << 20));      //  8 MiB  V [B,H,hd,S]
  _Float16* oh  = (_Float16*)(ws + (40ull << 20));      //  8 MiB  attn out [TOK,DIM]

  k_cvt_x<<<(TOK * DIM + 255) / 256, 256, 0, stream>>>(x, xh, TOK * DIM);
  k_transpose<<<(DIM * 3 * DIM + 255) / 256, 256, 0, stream>>>(Wqkv, wqt, DIM, 3 * DIM);
  k_transpose<<<(DIM * DIM + 255) / 256, 256, 0, stream>>>(Wout, wot, DIM, DIM);

  k_qkv<<<(TOK / 32) * (3 * DIM / 64) / 4, 128, 0, stream>>>(xh, wqt, bqkv, qh, kh, vt);
  k_attn<<<dim3(NB * NH, SEQ / 128), 128, 0, stream>>>(qh, kh, vt, oh);
  k_proj<<<(TOK / 32) * (DIM / 64) / 4, 128, 0, stream>>>(oh, wot, bout, (float*)d_out);
}
